// ConditionalLatentTrajectoryGenerator_240518168845
// MI455X (gfx1250) — matrix operand order repaired and checked by reading, not compiled, not run on hardware
//
#include <hip/hip_runtime.h>
#include <hip/hip_bf16.h>

// ---------------------------------------------------------------------------
// ConditionalLatentTrajectoryGenerator for MI455X (gfx1250, wave32, WMMA)
// bf16 WMMA + f32 accumulation. GRU weight (B) tiles staged through LDS with
// double-buffered global_load_async_to_lds_b128 (ASYNCcnt) to remove the 8x
// per-wave duplication of weight traffic; activations (A) loaded directly.
// ---------------------------------------------------------------------------

typedef __attribute__((ext_vector_type(16))) __bf16 v16bf;
typedef __attribute__((ext_vector_type(8)))  __bf16 v8bf;
typedef __attribute__((ext_vector_type(8)))  float  v8f;

#define HD   1024   // hidden dim
#define LDIM 256    // latent dim
#define CD   256    // cond dim
#define BB   128    // batch
#define TT   512    // rollout length (max_len; fixed by harness dims)
#define KX1  512    // L + C (GRU1 input width)

union V16U { v16bf v; v8bf h[2]; };

// A fragment (16x32 bf16, M x K) from row-major bf16 matrix.
// Layout: lane l -> row M = l&15; K = kb..kb+15 contiguous,
// kb = (l>>4)*16 (matches packed B).
__device__ __forceinline__ v16bf load_a_frag(const __bf16* base, int m0, int k0, int ld) {
  int l = threadIdx.x & 31;
  const __bf16* p = base + (size_t)(m0 + (l & 15)) * ld + k0 + ((l >> 4) << 4);
  V16U u;
  u.h[0] = *(const v8bf*)(p);
  u.h[1] = *(const v8bf*)(p + 8);
  return u.v;
}

// B fragment (32x16 bf16) from a packed 1KB tile (global memory).
__device__ __forceinline__ v16bf load_b_frag(const __bf16* packed, int tile) {
  int l = threadIdx.x & 31;
  const v8bf* p = (const v8bf*)(packed + ((size_t)tile * 32 + l) * 16);
  V16U u;
  u.h[0] = p[0];
  u.h[1] = p[1];
  return u.v;
}

// B fragment from a packed 1KB tile staged in LDS (-> ds_load_b128 x2).
__device__ __forceinline__ v16bf load_b_lds(const __bf16* sb) {
  int l = threadIdx.x & 31;
  const v8bf* p = (const v8bf*)(sb + l * 16);
  V16U u;
  u.h[0] = p[0];
  u.h[1] = p[1];
  return u.v;
}

__device__ __forceinline__ v8f wmma_bf16(v16bf a, v16bf b, v8f c) {
  // (neg_a, A, neg_b, B, c_mod, C, reuse_a, reuse_b)
  return __builtin_amdgcn_wmma_f32_16x16x32_bf16(false, a, false, b, (short)0, c, false, false);
}

__device__ __forceinline__ float sigm(float x) {
  return 1.0f / (1.0f + __expf(-x));
}

// Async copy of 16 bytes/lane from global to LDS (gfx1250 TDM-adjacent path,
// tracked with ASYNCcnt).
__device__ __forceinline__ void async_ld128(unsigned lds_addr, const void* gptr) {
  asm volatile("global_load_async_to_lds_b128 %0, %1, off"
               :: "v"(lds_addr), "v"((unsigned long long)(uintptr_t)gptr)
               : "memory");
}

// ---------------------------------------------------------------------------
// Phase 0: pack fp32 weight matrix [R x K] (row-major) into bf16 B-fragment
// order. B layout used here (contiguous k, same as A): lane l -> col N = l&15, K = kt*32 + (l>>4)*16 + 0..15.
// ---------------------------------------------------------------------------
__global__ void pack_w_kernel(const float* __restrict__ W, __bf16* __restrict__ P,
                              int R, int K) {
  int tile = blockIdx.x;           // tile = rt*(K/32) + kt
  int Kt = K >> 5;
  int rt = tile / Kt, kt = tile % Kt;
  int l = threadIdx.x;
  int r  = rt * 16 + (l & 15);
  int kb = kt * 32 + ((l >> 4) << 4);
  (void)R;
  __bf16* dst = P + ((size_t)tile * 32 + l) * 16;
  const float* src = W + (size_t)r * K + kb;
#pragma unroll
  for (int e = 0; e < 16; ++e) dst[e] = (__bf16)src[e];
}

// xcat[b][0:256] = z_start, xcat[b][256:512] = cond_emb (bf16)
__global__ void init_xcat_kernel(const float* __restrict__ z, const float* __restrict__ c,
                                 __bf16* __restrict__ xcat) {
  int i = blockIdx.x * 256 + threadIdx.x;     // BB*KX1 = 65536
  int b = i >> 9, col = i & 511;
  float v = (col < 256) ? z[b * 256 + col] : c[b * 256 + (col - 256)];
  xcat[i] = (__bf16)v;
}

// ---------------------------------------------------------------------------
// Generic setup matmul (one-time): out = A[BBxK] @ W^T + bias.
//  mode 0: relu -> bf16 ; mode 1: h0 -> both f32 states + bf16 copies ;
//  mode 2: FiLM -> scale = 1+gamma, beta (f32)
// ---------------------------------------------------------------------------
__global__ void __launch_bounds__(256) mm_setup_kernel(
    const __bf16* __restrict__ A, int lda, int K,
    const __bf16* __restrict__ Wp, const float* __restrict__ bias, int mode,
    __bf16* __restrict__ ob, float* __restrict__ o1, float* __restrict__ o2,
    __bf16* __restrict__ o3, __bf16* __restrict__ o4) {
  int wave = threadIdx.x >> 5, lane = threadIdx.x & 31;
  int m0 = wave * 16;
  int jt = blockIdx.x;
  int Kt = K >> 5;
  v8f acc = {};
  for (int kt = 0; kt < Kt; ++kt) {
    v16bf a = load_a_frag(A, m0, kt * 32, lda);
    v16bf b = load_b_frag(Wp, jt * Kt + kt);
    acc = wmma_bf16(a, b, acc);
  }
  int n = jt * 16 + (lane & 15);
  float bs = bias[n];
#pragma unroll
  for (int v = 0; v < 8; ++v) {
    int m = m0 + v + ((lane >> 4) << 3);    // C layout: M = vgpr + 8*(lane>=16)
    float val = acc[v] + bs;
    if (mode == 0) {
      ob[(size_t)m * HD + n] = (__bf16)fmaxf(val, 0.0f);
    } else if (mode == 1) {
      o1[(size_t)m * HD + n] = val;
      o2[(size_t)m * HD + n] = val;
      o3[(size_t)m * HD + n] = (__bf16)val;
      o4[(size_t)m * HD + n] = (__bf16)val;
    } else {
      if (n < HD) o1[(size_t)m * HD + n] = 1.0f + val;   // scale
      else        o2[(size_t)m * HD + (n - HD)] = val;   // beta
    }
  }
}

// ---------------------------------------------------------------------------
// Fused GRU cell step with async LDS staging of weight tiles.
// Grid = H/16 = 64 blocks (one 16-col N tile each), 8 waves = 8 M tiles over
// the 128 batch rows. Per (M,N) tile: 4 f32 accumulators (ir+hr, iz+hz, inn,
// hn). Per 32-wide K step: 3 x 1KB weight tiles (r/z/n) staged in LDS by
// waves 0-5 via global_load_async_to_lds_b128, double-buffered one K-step
// ahead; all waves consume via ds_load_b128 + 3 WMMAs.
// ---------------------------------------------------------------------------
__global__ void __launch_bounds__(256) gru_step_kernel(
    const __bf16* __restrict__ xact, int ldx, int Kx,   // input activations
    const __bf16* __restrict__ hact,                    // h_{t-1} bf16 [BB][HD]
    const float*  __restrict__ hin,                     // h_{t-1} f32
    float*        __restrict__ hout,                    // h_t f32
    const __bf16* __restrict__ wih, const __bf16* __restrict__ whh,
    const float*  __restrict__ bih, const float*  __restrict__ bhh,
    __bf16* __restrict__ hact_out,                      // h_t bf16
    const float* __restrict__ scale, const float* __restrict__ beta,
    __bf16* __restrict__ yact) {
  __shared__ __bf16 smem[2 * 3 * 512];     // 2 buffers x 3 gate tiles x 1KB

  int tid = threadIdx.x;
  int wave = tid >> 5, lane = tid & 31;
  int m0 = wave * 16;
  int jt = blockIdx.x;             // N tile within H
  const int GT = HD >> 4;          // row-tiles per gate = 64
  int Ktx = Kx >> 5, Kth = HD >> 5;
  int NK = Ktx + Kth;

  // Loader role: threads 0..191 (waves 0-5, uniform per wave). Each thread
  // copies one 16B chunk; thread i -> gate g = i>>6, chunk c = i&63.
  bool loader = (tid < 192);
  int lg = tid >> 6;
  int lc = tid & 63;

  auto issue = [&](int i, int buf) {
    if (!loader) return;
    const __bf16* base;
    int tidx;
    if (i < Ktx) { base = wih; tidx = (lg * GT + jt) * Ktx + i; }
    else         { base = whh; tidx = (lg * GT + jt) * Kth + (i - Ktx); }
    const __bf16* src = base + (size_t)tidx * 512 + lc * 8;
    unsigned dst = (unsigned)(uintptr_t)&smem[buf * 1536 + lg * 512 + lc * 8];
    async_ld128(dst, src);
  };

  v8f ar = {}, az = {}, an = {}, ah = {};
  issue(0, 0);
  for (int i = 0; i < NK; ++i) {
    int cur = i & 1;
    if (i + 1 < NK) {
      issue(i + 1, cur ^ 1);
      asm volatile("s_wait_asynccnt 0x1" ::: "memory");  // cur buffer landed
    } else {
      asm volatile("s_wait_asynccnt 0x0" ::: "memory");
    }
    __syncthreads();                                     // cross-wave visibility

    v16bf a = (i < Ktx) ? load_a_frag(xact, m0, i * 32, ldx)
                        : load_a_frag(hact, m0, (i - Ktx) * 32, HD);
    const __bf16* sb = &smem[cur * 1536];
    v16bf b0 = load_b_lds(sb);
    v16bf b1 = load_b_lds(sb + 512);
    v16bf b2 = load_b_lds(sb + 1024);
    ar = wmma_bf16(a, b0, ar);
    az = wmma_bf16(a, b1, az);
    if (i < Ktx) an = wmma_bf16(a, b2, an);   // inn
    else         ah = wmma_bf16(a, b2, ah);   // hn

    __syncthreads();   // all waves done reading cur before it is re-issued
  }

  int n = jt * 16 + (lane & 15);
  float brv  = bih[n]          + bhh[n];
  float bzv  = bih[HD + n]     + bhh[HD + n];
  float binv = bih[2 * HD + n];
  float bhnv = bhh[2 * HD + n];
#pragma unroll
  for (int v = 0; v < 8; ++v) {
    int m = m0 + v + ((lane >> 4) << 3);
    size_t idx = (size_t)m * HD + n;
    float r  = sigm(ar[v] + brv);
    float z  = sigm(az[v] + bzv);
    float nn = tanhf(an[v] + binv + r * (ah[v] + bhnv));
    float hn = (1.0f - z) * nn + z * hin[idx];
    hout[idx]     = hn;
    hact_out[idx] = (__bf16)hn;
    if (yact) yact[idx] = (__bf16)(scale[idx] * hn + beta[idx]);
  }
}

// ---------------------------------------------------------------------------
// Head: next_z = y @ head_w^T + head_b. Writes out[b][t][l] (f32) and the
// next-step latent into xcat cols 0..255 (bf16). Grid = L/16 = 16 blocks.
// ---------------------------------------------------------------------------
__global__ void __launch_bounds__(256) head_step_kernel(
    const __bf16* __restrict__ yact, const __bf16* __restrict__ wp,
    const float* __restrict__ bias, float* __restrict__ out, int t,
    __bf16* __restrict__ xcat) {
  int wave = threadIdx.x >> 5, lane = threadIdx.x & 31;
  int m0 = wave * 16;
  int jt = blockIdx.x;
  const int Kt = HD >> 5;
  v8f acc = {};
  for (int kt = 0; kt < Kt; ++kt) {
    v16bf a = load_a_frag(yact, m0, kt * 32, HD);
    v16bf b = load_b_frag(wp, jt * Kt + kt);
    acc = wmma_bf16(a, b, acc);
  }
  int n = jt * 16 + (lane & 15);
  float bs = bias[n];
#pragma unroll
  for (int v = 0; v < 8; ++v) {
    int m = m0 + v + ((lane >> 4) << 3);
    float val = acc[v] + bs;
    out[(size_t)m * TT * LDIM + (size_t)t * LDIM + n] = val;
    xcat[(size_t)m * KX1 + n] = (__bf16)val;
  }
}

// ---------------------------------------------------------------------------
extern "C" void kernel_launch(void* const* d_in, const int* in_sizes, int n_in,
                              void* d_out, int out_size, void* d_ws, size_t ws_size,
                              hipStream_t stream) {
  (void)in_sizes; (void)n_in; (void)out_size; (void)ws_size;
  const float* z_start = (const float*)d_in[0];
  const float* cond    = (const float*)d_in[1];
  const float* z2h_w1  = (const float*)d_in[3];
  const float* z2h_b1  = (const float*)d_in[4];
  const float* z2h_w2  = (const float*)d_in[5];
  const float* z2h_b2  = (const float*)d_in[6];
  const float* w_ih1   = (const float*)d_in[7];
  const float* w_hh1   = (const float*)d_in[8];
  const float* b_ih1   = (const float*)d_in[9];
  const float* b_hh1   = (const float*)d_in[10];
  const float* w_ih2   = (const float*)d_in[11];
  const float* w_hh2   = (const float*)d_in[12];
  const float* b_ih2   = (const float*)d_in[13];
  const float* b_hh2   = (const float*)d_in[14];
  const float* film_w  = (const float*)d_in[15];
  const float* film_b  = (const float*)d_in[16];
  const float* head_w  = (const float*)d_in[17];
  const float* head_b  = (const float*)d_in[18];
  float* out = (float*)d_out;

  // ---- workspace layout (256B aligned) ----
  char* p = (char*)d_ws;
  auto alloc = [&](size_t bytes) -> void* {
    void* r = (void*)p;
    p += (bytes + 255) & ~(size_t)255;
    return r;
  };
  __bf16* pw_z1   = (__bf16*)alloc((size_t)1024 * 256  * 2);
  __bf16* pw_z2   = (__bf16*)alloc((size_t)1024 * 1024 * 2);
  __bf16* pw_ih1  = (__bf16*)alloc((size_t)3072 * 512  * 2);
  __bf16* pw_hh1  = (__bf16*)alloc((size_t)3072 * 1024 * 2);
  __bf16* pw_ih2  = (__bf16*)alloc((size_t)3072 * 1024 * 2);
  __bf16* pw_hh2  = (__bf16*)alloc((size_t)3072 * 1024 * 2);
  __bf16* pw_film = (__bf16*)alloc((size_t)2048 * 256  * 2);
  __bf16* pw_head = (__bf16*)alloc((size_t)256  * 1024 * 2);
  __bf16* xcat    = (__bf16*)alloc((size_t)BB * KX1 * 2);
  __bf16* tmp_bf  = (__bf16*)alloc((size_t)BB * HD * 2);
  float*  h1s[2]  = { (float*)alloc((size_t)BB * HD * 4), (float*)alloc((size_t)BB * HD * 4) };
  float*  h2s[2]  = { (float*)alloc((size_t)BB * HD * 4), (float*)alloc((size_t)BB * HD * 4) };
  __bf16* h1a[2]  = { (__bf16*)alloc((size_t)BB * HD * 2), (__bf16*)alloc((size_t)BB * HD * 2) };
  __bf16* h2a[2]  = { (__bf16*)alloc((size_t)BB * HD * 2), (__bf16*)alloc((size_t)BB * HD * 2) };
  float*  scl     = (float*)alloc((size_t)BB * HD * 4);
  float*  bta     = (float*)alloc((size_t)BB * HD * 4);
  __bf16* ya      = (__bf16*)alloc((size_t)BB * HD * 2);

  // ---- Phase 0: pack weights (one-time; bf16 copies then live in L2) ----
  auto pack = [&](const float* w, __bf16* dst, int R, int K) {
    pack_w_kernel<<<(R / 16) * (K / 32), 32, 0, stream>>>(w, dst, R, K);
  };
  pack(z2h_w1, pw_z1,   1024, 256);
  pack(z2h_w2, pw_z2,   1024, 1024);
  pack(w_ih1,  pw_ih1,  3072, 512);
  pack(w_hh1,  pw_hh1,  3072, 1024);
  pack(w_ih2,  pw_ih2,  3072, 1024);
  pack(w_hh2,  pw_hh2,  3072, 1024);
  pack(film_w, pw_film, 2048, 256);
  pack(head_w, pw_head, 256,  1024);

  init_xcat_kernel<<<(BB * KX1) / 256, 256, 0, stream>>>(z_start, cond, xcat);

  // h0 = relu(z @ z2h_w1^T + b1) @ z2h_w2^T + b2 ; FiLM params
  mm_setup_kernel<<<HD / 16, 256, 0, stream>>>(
      xcat, KX1, 256, pw_z1, z2h_b1, 0, tmp_bf, nullptr, nullptr, nullptr, nullptr);
  mm_setup_kernel<<<HD / 16, 256, 0, stream>>>(
      tmp_bf, HD, HD, pw_z2, z2h_b2, 1, nullptr, h1s[0], h2s[0], h1a[0], h2a[0]);
  mm_setup_kernel<<<(2 * HD) / 16, 256, 0, stream>>>(
      xcat + 256, KX1, 256, pw_film, film_b, 2, nullptr, scl, bta, nullptr, nullptr);

  // ---- rollout: T sequential steps, 3 kernels each ----
  for (int t = 0; t < TT; ++t) {
    int s = t & 1, d = s ^ 1;
    // GRU layer 1: x_in = xcat [BB x 512]
    gru_step_kernel<<<HD / 16, 256, 0, stream>>>(
        xcat, KX1, KX1, h1a[s], h1s[s], h1s[d],
        pw_ih1, pw_hh1, b_ih1, b_hh1, h1a[d],
        nullptr, nullptr, nullptr);
    // GRU layer 2 (+ FiLM epilogue -> y)
    gru_step_kernel<<<HD / 16, 256, 0, stream>>>(
        h1a[d], HD, HD, h2a[s], h2s[s], h2s[d],
        pw_ih2, pw_hh2, b_ih2, b_hh2, h2a[d],
        scl, bta, ya);
    // head -> out[:, t, :] and next latent into xcat cols 0..255
    head_step_kernel<<<LDIM / 16, 256, 0, stream>>>(
        ya, pw_head, head_b, out, t, xcat);
  }
}
